// MultiAtt_58626303590888
// MI455X (gfx1250) — matrix operand order repaired and checked by reading, not compiled, not run on hardware
//
#include <hip/hip_runtime.h>
#include <hip/hip_bf16.h>
#include <stdint.h>

// ---------------- problem constants ----------------
#define B_    64
#define SP_   256
#define SH_   256
#define D_    600
#define H_    300
#define DP_   608            // D padded to multiple of 32 (WMMA K)
#define ENHK_ (4*DP_)        // 2432, padded enhanced-feature K
#define HP_   320            // H padded to multiple of 32 (WMMA N blocking)
#define NEG_  (-999999.0f)

// ---------------- WMMA types ----------------
typedef __bf16 bf_t;
typedef bf_t  v16bf __attribute__((ext_vector_type(16)));
typedef float v8f   __attribute__((ext_vector_type(8)));

struct __attribute__((aligned(16))) Q4 { unsigned x, y, z, w; };
struct __attribute__((aligned(32))) Q8 { Q4 lo, hi; };

__device__ __forceinline__ v16bf as_v16bf(Q8 q) {
  return __builtin_bit_cast(v16bf, q);
}

__device__ __forceinline__ unsigned short f2bf(float x) {
  unsigned u = __float_as_uint(x);
  return (unsigned short)((u + 0x7fffu + ((u >> 16) & 1u)) >> 16); // RNE
}
__device__ __forceinline__ float bf2f(unsigned short s) {
  return __uint_as_float(((unsigned)s) << 16);
}
__device__ __forceinline__ unsigned pack2bf(float a, float b) {
  return (unsigned)f2bf(a) | ((unsigned)f2bf(b) << 16);
}
__device__ __forceinline__ unsigned bfpair(unsigned a, unsigned b, bool mul) {
  float a0 = bf2f((unsigned short)a), a1 = bf2f((unsigned short)(a >> 16));
  float b0 = bf2f((unsigned short)b), b1 = bf2f((unsigned short)(b >> 16));
  float r0 = mul ? a0 * b0 : a0 - b0;
  float r1 = mul ? a1 * b1 : a1 - b1;
  return (unsigned)f2bf(r0) | ((unsigned)f2bf(r1) << 16);
}
__device__ __forceinline__ Q4 q4op(Q4 a, Q4 b, bool mul) {
  Q4 r;
  r.x = bfpair(a.x, b.x, mul); r.y = bfpair(a.y, b.y, mul);
  r.z = bfpair(a.z, b.z, mul); r.w = bfpair(a.w, b.w, mul);
  return r;
}

#define LDQ4(p) (*(const Q4*)(p))

__device__ __forceinline__ v8f wmma_bf16(Q8 a, Q8 b, v8f c) {
  return __builtin_amdgcn_wmma_f32_16x16x32_bf16(
      false, as_v16bf(a), false, as_v16bf(b), (short)0, c, false, false);
}

// ---------------- Tensor Data Mover staging ----------------
#if __has_builtin(__builtin_amdgcn_tensor_load_to_lds) && __has_builtin(__builtin_amdgcn_s_wait_tensorcnt)
#define HAVE_TDM 1
typedef unsigned int u32x4 __attribute__((ext_vector_type(4)));
typedef int i32x4 __attribute__((ext_vector_type(4)));
typedef int i32x8 __attribute__((ext_vector_type(8)));

// Load a contiguous [32 rows][256 bf16] tile (row stride 256 elems) into LDS.
// D# bitfields per CDNA5 ISA:
//   group0: count=1 | lds_addr | global_addr[56:0] | type=2
//   group1: dw0 = workgroup_mask(0) | data_size=1 (2 bytes)
//           dw1[31:16] = tensor_dim0 lo16 (256)
//           dw2[15:0]  = tensor_dim0 hi16 (0), dw2[31:16] = tensor_dim1 lo16 (32)
//           dw3[15:0]  = tensor_dim1 hi16 (0), dw3[31:16] = tile_dim0 (256)
//           dw4[15:0]  = tile_dim1 (32), dw4[31:16] = tile_dim2 (0)
//           dw5        = tensor_dim0_stride lo32 (256)
//           dw6/dw7    = stride hi / tensor_dim1_stride (unused -> 0)
__device__ __forceinline__ void tdm_load_tile_32x256_bf16(const unsigned short* gsrc,
                                                          unsigned lds_off) {
  unsigned long long ga = (unsigned long long)(const void*)gsrc;
  u32x4 g0;
  g0[0] = 1u;                                             // count=1 (valid user D#)
  g0[1] = lds_off;                                        // lds_addr
  g0[2] = (unsigned)(ga & 0xffffffffull);                 // global_addr[31:0]
  g0[3] = (unsigned)((ga >> 32) & 0x1ffffffull)           // global_addr[56:32]
        | (2u << 30);                                     // type=2 ("image")
  i32x8 g1;
  g1[0] = 0x00010000;            // workgroup_mask=0, data_size=1 (2 bytes)
  g1[1] = (int)(256u << 16);     // tensor_dim0 = 256 (low 16 in dw1[31:16])
  g1[2] = (int)(32u << 16);      // td0 hi16 = 0, tensor_dim1 = 32 (dw2[31:16])
  g1[3] = (int)(256u << 16);     // td1 hi16 = 0, tile_dim0 = 256 (dw3[31:16])
  g1[4] = 32;                    // tile_dim1 = 32, tile_dim2 = 0
  g1[5] = 256;                   // tensor_dim0_stride = 256 (low 32)
  g1[6] = 0;                     // stride hi16, tensor_dim1_stride lo16 (unused)
  g1[7] = 0;
  i32x4 z4 = (i32x4)0;
  i32x8 z8 = (i32x8)0;
  __builtin_amdgcn_tensor_load_to_lds(g0, g1, z4, z4, z8, 0);
}
#endif

// ---------------- conversion / staging kernels ----------------
__global__ void cvt_p_kernel(const float* __restrict__ p, const float* __restrict__ wt,
                             unsigned short* __restrict__ p_bf,
                             unsigned short* __restrict__ pwd_bf,
                             unsigned short* __restrict__ pt_bf,
                             float* __restrict__ r) {
  int b = blockIdx.y, s = blockIdx.x, tid = threadIdx.x;
  const float* wa = wt;
  const float* wc = wt + 2 * D_;
  const float* wd = wt + 3 * D_;
  const float* row = p + ((size_t)b * SP_ + s) * D_;
  size_t ro = ((size_t)b * SP_ + s) * DP_;
  float acc = 0.f;
  for (int d = tid; d < DP_; d += 256) {
    float v = (d < D_) ? row[d] : 0.f;
    unsigned short bv = f2bf(v);
    p_bf[ro + d] = bv;
    float wdv = (d < D_) ? wd[d] : 0.f;
    pwd_bf[ro + d] = f2bf(v * wdv);
    pt_bf[((size_t)b * DP_ + d) * SP_ + s] = bv;
    if (d < D_) acc += v * (wa[d] + wc[d]);
  }
  __shared__ float red[256];
  red[tid] = acc; __syncthreads();
  for (int ss = 128; ss > 0; ss >>= 1) {
    if (tid < ss) red[tid] += red[tid + ss];
    __syncthreads();
  }
  if (tid == 0) r[b * SP_ + s] = red[0];
}

__global__ void cvt_h_kernel(const float* __restrict__ h, const float* __restrict__ wt,
                             unsigned short* __restrict__ h_bf,
                             unsigned short* __restrict__ ht_bf,
                             float* __restrict__ c) {
  int b = blockIdx.y, s = blockIdx.x, tid = threadIdx.x;
  const float* wb = wt + D_;
  const float* wc = wt + 2 * D_;
  const float* row = h + ((size_t)b * SH_ + s) * D_;
  size_t ro = ((size_t)b * SH_ + s) * DP_;
  float acc = 0.f;
  for (int d = tid; d < DP_; d += 256) {
    float v = (d < D_) ? row[d] : 0.f;
    unsigned short bv = f2bf(v);
    h_bf[ro + d] = bv;
    ht_bf[((size_t)b * DP_ + d) * SH_ + s] = bv;
    if (d < D_) acc += v * (wb[d] - wc[d]);
  }
  __shared__ float red[256];
  red[tid] = acc; __syncthreads();
  for (int ss = 128; ss > 0; ss >>= 1) {
    if (tid < ss) red[tid] += red[tid + ss];
    __syncthreads();
  }
  if (tid == 0) c[b * SH_ + s] = red[0];
}

__global__ void cvt_w_kernel(const float* __restrict__ W, unsigned short* __restrict__ Wt) {
  int idx = blockIdx.x * 256 + threadIdx.x;
  if (idx >= HP_ * ENHK_) return;
  int hh = idx / ENHK_, k = idx - hh * ENHK_;
  unsigned short v = 0;
  if (hh < H_) {
    int sec = k / DP_, kk = k - sec * DP_;
    if (kk < D_) v = f2bf(W[(size_t)(sec * D_ + kk) * H_ + hh]);
  }
  Wt[idx] = v;
}

// -------- score GEMM: S = (p*wd) @ h^T, 32x32 per wave (2x2 WMMA blocking) --------
__global__ void score_kernel(const unsigned short* __restrict__ pwd_bf,
                             const unsigned short* __restrict__ h_bf,
                             float* __restrict__ S) {
  int b = blockIdx.z;
  int lane = threadIdx.x & 31, wave = threadIdx.x >> 5;
  int half = lane >> 4, lo = lane & 15;
  int m0 = blockIdx.x * 64 + (wave & 1) * 32;
  int n0 = blockIdx.y * 128 + (wave >> 1) * 32;
  const unsigned short* Ar0 = pwd_bf + ((size_t)b * SP_ + m0 + lo) * DP_;
  const unsigned short* Ar1 = Ar0 + (size_t)16 * DP_;
  const unsigned short* Br0 = h_bf + ((size_t)b * SH_ + n0 + lo) * DP_;
  const unsigned short* Br1 = Br0 + (size_t)16 * DP_;
  v8f a00 = {}, a01 = {}, a10 = {}, a11 = {};
  for (int k0 = 0; k0 < DP_; k0 += 32) {
    Q8 aq0, aq1, bq0, bq1;
    aq0.lo = LDQ4(Ar0 + k0 + 8 * half); aq0.hi = LDQ4(Ar0 + k0 + 16 + 8 * half);
    aq1.lo = LDQ4(Ar1 + k0 + 8 * half); aq1.hi = LDQ4(Ar1 + k0 + 16 + 8 * half);
    bq0.lo = LDQ4(Br0 + k0 + 8 * half); bq0.hi = LDQ4(Br0 + k0 + 16 + 8 * half);
    bq1.lo = LDQ4(Br1 + k0 + 8 * half); bq1.hi = LDQ4(Br1 + k0 + 16 + 8 * half);
    a00 = wmma_bf16(aq0, bq0, a00); a01 = wmma_bf16(aq0, bq1, a01);
    a10 = wmma_bf16(aq1, bq0, a10); a11 = wmma_bf16(aq1, bq1, a11);
  }
  float* Sb = S + (size_t)b * SP_ * SH_;
#pragma unroll
  for (int v = 0; v < 8; ++v) {
    int r0 = m0 + v + 8 * half, r1 = r0 + 16;
    Sb[(size_t)r0 * SH_ + n0 + lo]      = a00[v];
    Sb[(size_t)r0 * SH_ + n0 + 16 + lo] = a01[v];
    Sb[(size_t)r1 * SH_ + n0 + lo]      = a10[v];
    Sb[(size_t)r1 * SH_ + n0 + 16 + lo] = a11[v];
  }
}

// ---------------- softmax over axis 2 (rows) -> Aa bf16 ----------------
__global__ void softmax_row_kernel(const float* __restrict__ S, const float* __restrict__ r,
                                   const float* __restrict__ c, const int* __restrict__ plen,
                                   unsigned short* __restrict__ Aa) {
  int b = blockIdx.y;
  int wave = threadIdx.x >> 5, lane = threadIdx.x & 31;
  int p = blockIdx.x * 8 + wave;
  const float* row = S + ((size_t)b * SP_ + p) * SH_;
  const float* cb = c + b * SH_;
  float bias = r[b * SP_ + p] + ((p < plen[b]) ? 0.f : NEG_);
  float4 s0 = *(const float4*)(row + lane * 8);
  float4 s1 = *(const float4*)(row + lane * 8 + 4);
  float4 c0 = *(const float4*)(cb + lane * 8);
  float4 c1 = *(const float4*)(cb + lane * 8 + 4);
  float vals[8];
  vals[0] = s0.x + c0.x + bias; vals[1] = s0.y + c0.y + bias;
  vals[2] = s0.z + c0.z + bias; vals[3] = s0.w + c0.w + bias;
  vals[4] = s1.x + c1.x + bias; vals[5] = s1.y + c1.y + bias;
  vals[6] = s1.z + c1.z + bias; vals[7] = s1.w + c1.w + bias;
  float m = vals[0];
#pragma unroll
  for (int j = 1; j < 8; ++j) m = fmaxf(m, vals[j]);
  for (int off = 16; off > 0; off >>= 1) m = fmaxf(m, __shfl_xor(m, off, 32));
  float sum = 0.f;
#pragma unroll
  for (int j = 0; j < 8; ++j) { vals[j] = __expf(vals[j] - m); sum += vals[j]; }
  for (int off = 16; off > 0; off >>= 1) sum += __shfl_xor(sum, off, 32);
  float inv = 1.0f / sum;
  Q4 o;
  o.x = pack2bf(vals[0] * inv, vals[1] * inv);
  o.y = pack2bf(vals[2] * inv, vals[3] * inv);
  o.z = pack2bf(vals[4] * inv, vals[5] * inv);
  o.w = pack2bf(vals[6] * inv, vals[7] * inv);
  *(Q4*)(Aa + ((size_t)b * SP_ + p) * SH_ + lane * 8) = o;
}

// -------- softmax over axis 1 (columns) -> Abt bf16 [B][SH][SP] --------
__global__ void softmax_col_kernel(const float* __restrict__ S, const float* __restrict__ r,
                                   const float* __restrict__ c, const int* __restrict__ hlen,
                                   unsigned short* __restrict__ Abt) {
  int b = blockIdx.y;
  int wave = threadIdx.x >> 5, lane = threadIdx.x & 31;
  int hcol = blockIdx.x * 8 + wave;
  const float* Sb = S + (size_t)b * SP_ * SH_;
  const float* rb = r + b * SP_;
  float bias = c[b * SH_ + hcol] + ((hcol < hlen[b]) ? 0.f : NEG_);
  float4 r0 = *(const float4*)(rb + lane * 8);
  float4 r1 = *(const float4*)(rb + lane * 8 + 4);
  float vals[8];
#pragma unroll
  for (int j = 0; j < 8; ++j) vals[j] = Sb[(size_t)(lane * 8 + j) * SH_ + hcol];
  vals[0] += r0.x + bias; vals[1] += r0.y + bias;
  vals[2] += r0.z + bias; vals[3] += r0.w + bias;
  vals[4] += r1.x + bias; vals[5] += r1.y + bias;
  vals[6] += r1.z + bias; vals[7] += r1.w + bias;
  float m = vals[0];
#pragma unroll
  for (int j = 1; j < 8; ++j) m = fmaxf(m, vals[j]);
  for (int off = 16; off > 0; off >>= 1) m = fmaxf(m, __shfl_xor(m, off, 32));
  float sum = 0.f;
#pragma unroll
  for (int j = 0; j < 8; ++j) { vals[j] = __expf(vals[j] - m); sum += vals[j]; }
  for (int off = 16; off > 0; off >>= 1) sum += __shfl_xor(sum, off, 32);
  float inv = 1.0f / sum;
  Q4 o;
  o.x = pack2bf(vals[0] * inv, vals[1] * inv);
  o.y = pack2bf(vals[2] * inv, vals[3] * inv);
  o.z = pack2bf(vals[4] * inv, vals[5] * inv);
  o.w = pack2bf(vals[6] * inv, vals[7] * inv);
  *(Q4*)(Abt + ((size_t)b * SH_ + hcol) * SP_ + lane * 8) = o;
}

// ------- attended GEMMs: ap = Aa @ h (via ht), ah = Abt @ p (via pt) -------
// B tile (32 rows x 256 bf16 = 16 KB) staged to LDS by the Tensor Data Mover,
// shared by all 8 waves; each wave computes a 32x32 output block.
__global__ void attended_kernel(const unsigned short* __restrict__ Aa,
                                const unsigned short* __restrict__ Abt,
                                const unsigned short* __restrict__ ht,
                                const unsigned short* __restrict__ pt,
                                unsigned short* __restrict__ ap,
                                unsigned short* __restrict__ ah) {
  __shared__ __align__(16) unsigned short Bs[32 * 256];
  int b = blockIdx.z >> 1, which = blockIdx.z & 1;
  int lane = threadIdx.x & 31, wave = threadIdx.x >> 5;
  int half = lane >> 4, lo = lane & 15;
  int m0 = wave * 32;
  int n0 = blockIdx.x * 32;
  const unsigned short* A  = which ? Abt : Aa;
  const unsigned short* Bm = which ? pt  : ht;
  unsigned short* out = which ? ah : ap;

  const unsigned short* gB = Bm + ((size_t)b * DP_ + n0) * 256;
#ifdef HAVE_TDM
  if (wave == 0) {
    tdm_load_tile_32x256_bf16(gB, (unsigned)(size_t)(void*)Bs);
    __builtin_amdgcn_s_wait_tensorcnt(0);
  }
  __syncthreads();
#else
  for (int i = threadIdx.x; i < (32 * 256) / 8; i += 256)
    ((Q4*)Bs)[i] = ((const Q4*)gB)[i];
  __syncthreads();
#endif

  const unsigned short* Ar0 = A + ((size_t)b * 256 + m0 + lo) * 256;
  const unsigned short* Ar1 = Ar0 + 16 * 256;
  const unsigned short* Bl0 = Bs + (size_t)lo * 256;
  const unsigned short* Bl1 = Bs + (size_t)(16 + lo) * 256;
  v8f a00 = {}, a01 = {}, a10 = {}, a11 = {};
  for (int k0 = 0; k0 < 256; k0 += 32) {
    Q8 aq0, aq1, bq0, bq1;
    aq0.lo = LDQ4(Ar0 + k0 + 8 * half); aq0.hi = LDQ4(Ar0 + k0 + 16 + 8 * half);
    aq1.lo = LDQ4(Ar1 + k0 + 8 * half); aq1.hi = LDQ4(Ar1 + k0 + 16 + 8 * half);
    bq0.lo = LDQ4(Bl0 + k0 + 8 * half); bq0.hi = LDQ4(Bl0 + k0 + 16 + 8 * half);
    bq1.lo = LDQ4(Bl1 + k0 + 8 * half); bq1.hi = LDQ4(Bl1 + k0 + 16 + 8 * half);
    a00 = wmma_bf16(aq0, bq0, a00); a01 = wmma_bf16(aq0, bq1, a01);
    a10 = wmma_bf16(aq1, bq0, a10); a11 = wmma_bf16(aq1, bq1, a11);
  }
#pragma unroll
  for (int v = 0; v < 8; ++v) {
    int r0 = m0 + v + 8 * half, r1 = r0 + 16;
    size_t base = (size_t)b * 256;
    out[(base + r0) * DP_ + n0 + lo]      = f2bf(a00[v]);
    out[(base + r0) * DP_ + n0 + 16 + lo] = f2bf(a01[v]);
    out[(base + r1) * DP_ + n0 + lo]      = f2bf(a10[v]);
    out[(base + r1) * DP_ + n0 + 16 + lo] = f2bf(a11[v]);
  }
}

// ------- projection: relu([x, a, x-a, x*a] @ W + b), enh built on the fly -------
__device__ __forceinline__ Q8 enh_chunk(const unsigned short* brow, const unsigned short* arow,
                                        int sec, int c0, int c1) {
  Q8 aq;
  if (sec == 0) {
    aq.lo = LDQ4(brow + c0); aq.hi = LDQ4(brow + c1);
  } else if (sec == 1) {
    aq.lo = LDQ4(arow + c0); aq.hi = LDQ4(arow + c1);
  } else {
    Q4 x0 = LDQ4(brow + c0), x1 = LDQ4(brow + c1);
    Q4 a0 = LDQ4(arow + c0), a1 = LDQ4(arow + c1);
    bool mul = (sec == 3);
    aq.lo = q4op(x0, a0, mul); aq.hi = q4op(x1, a1, mul);
  }
  return aq;
}

__global__ void proj_kernel(const unsigned short* __restrict__ p_bf,
                            const unsigned short* __restrict__ h_bf,
                            const unsigned short* __restrict__ ap,
                            const unsigned short* __restrict__ ah,
                            const unsigned short* __restrict__ Wt,
                            const float* __restrict__ bias,
                            float* __restrict__ out) {
  int b = blockIdx.z >> 1, part = blockIdx.z & 1;
  int lane = threadIdx.x & 31, wave = threadIdx.x >> 5;
  int half = lane >> 4, lo = lane & 15;
  int m0 = wave * 32;
  int n0 = blockIdx.x * 32;
  const unsigned short* base = part ? h_bf : p_bf;
  const unsigned short* att  = part ? ah : ap;
  const unsigned short* br0 = base + ((size_t)b * 256 + m0 + lo) * DP_;
  const unsigned short* ar0 = att  + ((size_t)b * 256 + m0 + lo) * DP_;
  const unsigned short* br1 = br0 + (size_t)16 * DP_;
  const unsigned short* ar1 = ar0 + (size_t)16 * DP_;
  const unsigned short* wr0 = Wt + (size_t)(n0 + lo) * ENHK_;
  const unsigned short* wr1 = wr0 + (size_t)16 * ENHK_;
  v8f a00 = {}, a01 = {}, a10 = {}, a11 = {};
  for (int k0 = 0; k0 < ENHK_; k0 += 32) {
    int sec = k0 / DP_;
    int kk = k0 - sec * DP_;
    int c0 = kk + 8 * half, c1 = kk + 16 + 8 * half;
    Q8 aq0 = enh_chunk(br0, ar0, sec, c0, c1);
    Q8 aq1 = enh_chunk(br1, ar1, sec, c0, c1);
    Q8 bq0, bq1;
    bq0.lo = LDQ4(wr0 + k0 + 8 * half); bq0.hi = LDQ4(wr0 + k0 + 16 + 8 * half);
    bq1.lo = LDQ4(wr1 + k0 + 8 * half); bq1.hi = LDQ4(wr1 + k0 + 16 + 8 * half);
    a00 = wmma_bf16(aq0, bq0, a00); a01 = wmma_bf16(aq0, bq1, a01);
    a10 = wmma_bf16(aq1, bq0, a10); a11 = wmma_bf16(aq1, bq1, a11);
  }
  size_t obase = (size_t)b * (SP_ + SH_) + part * SP_;
#pragma unroll
  for (int nj = 0; nj < 2; ++nj) {
    int hh = n0 + nj * 16 + lo;
    if (hh < H_) {
      float bv = bias[hh];
#pragma unroll
      for (int v = 0; v < 8; ++v) {
        int r0 = m0 + v + 8 * half, r1 = r0 + 16;
        float v0 = (nj ? a01[v] : a00[v]) + bv;
        float v1 = (nj ? a11[v] : a10[v]) + bv;
        out[(obase + r0) * H_ + hh] = fmaxf(v0, 0.f);
        out[(obase + r1) * H_ + hh] = fmaxf(v1, 0.f);
      }
    }
  }
}

// ---------------- host launch ----------------
extern "C" void kernel_launch(void* const* d_in, const int* in_sizes, int n_in,
                              void* d_out, int out_size, void* d_ws, size_t ws_size,
                              hipStream_t stream) {
  (void)in_sizes; (void)n_in; (void)out_size; (void)ws_size;
  const float* p  = (const float*)d_in[0];
  const float* h  = (const float*)d_in[1];
  const float* wt = (const float*)d_in[2];
  const float* W  = (const float*)d_in[3];
  const float* bj = (const float*)d_in[4];
  const int* plen = (const int*)d_in[5];
  const int* hlen = (const int*)d_in[6];
  float* out = (float*)d_out;

  char* ws = (char*)d_ws;
  size_t off = 0;
  auto alloc = [&](size_t bytes) -> char* {
    char* r = ws + off;
    off = (off + bytes + 255) & ~(size_t)255;
    return r;
  };
  const size_t bsd = (size_t)B_ * 256 * DP_ * 2;   // bf16 [B][256][DP]
  unsigned short* p_bf = (unsigned short*)alloc(bsd);
  unsigned short* pwd  = (unsigned short*)alloc(bsd);
  unsigned short* pt   = (unsigned short*)alloc(bsd);
  unsigned short* h_bf = (unsigned short*)alloc(bsd);
  unsigned short* ht   = (unsigned short*)alloc(bsd);
  unsigned short* apb  = (unsigned short*)alloc(bsd);
  unsigned short* ahb  = (unsigned short*)alloc(bsd);
  float* S             = (float*)alloc((size_t)B_ * SP_ * SH_ * 4);
  unsigned short* Aa   = (unsigned short*)alloc((size_t)B_ * SP_ * SH_ * 2);
  unsigned short* Abt  = (unsigned short*)alloc((size_t)B_ * SP_ * SH_ * 2);
  unsigned short* Wt   = (unsigned short*)alloc((size_t)HP_ * ENHK_ * 2);
  float* r             = (float*)alloc((size_t)B_ * SP_ * 4);
  float* c             = (float*)alloc((size_t)B_ * SH_ * 4);

  cvt_p_kernel<<<dim3(SP_, B_), 256, 0, stream>>>(p, wt, p_bf, pwd, pt, r);
  cvt_h_kernel<<<dim3(SH_, B_), 256, 0, stream>>>(h, wt, h_bf, ht, c);
  cvt_w_kernel<<<(HP_ * ENHK_ + 255) / 256, 256, 0, stream>>>(W, Wt);

  score_kernel<<<dim3(SP_ / 64, SH_ / 128, B_), 256, 0, stream>>>(pwd, h_bf, S);

  softmax_row_kernel<<<dim3(SP_ / 8, B_), 256, 0, stream>>>(S, r, c, plen, Aa);
  softmax_col_kernel<<<dim3(SH_ / 8, B_), 256, 0, stream>>>(S, r, c, hlen, Abt);

  attended_kernel<<<dim3(DP_ / 32, 1, B_ * 2), 256, 0, stream>>>(Aa, Abt, ht, pt, apb, ahb);

  proj_kernel<<<dim3(HP_ / 32, 1, B_ * 2), 256, 0, stream>>>(p_bf, h_bf, apb, ahb, Wt, bj, out);
}
